// SoftDecisionTree_5781025980743
// MI455X (gfx1250) — matrix operand order repaired and checked by reading, not compiled, not run on hardware
//
#include <hip/hip_runtime.h>
#include <hip/hip_bf16.h>

typedef _Float16 h8   __attribute__((ext_vector_type(8)));
typedef _Float16 v16h __attribute__((ext_vector_type(16)));
typedef float    v8f  __attribute__((ext_vector_type(8)));

#define INPUT_DIM 64
#define N_NODES   127
#define N_LEAVES  128

// 128 threads = 4 wave32; each wave owns 32 batch rows; block owns 128 rows.
__global__ __launch_bounds__(128)
void soft_tree_kernel(const float* __restrict__ x,
                      const float* __restrict__ nw,
                      const float* __restrict__ nb,
                      const float* __restrict__ lv,
                      float* __restrict__ out,
                      int batch)
{
    __shared__ __align__(16) _Float16 s_wh[128 * 64];      // 16 KB  W as f16, row-major [n][k], n=127 zero pad
    __shared__ float s_bias[128];
    __shared__ float s_lv[128];
    __shared__ __align__(16) _Float16 s_xh[4][32 * 64];    // 16 KB  per-wave x tile as f16
    __shared__ float s_probs[4][32 * 128];                 // 64 KB  per-wave sigmoid probs

    const int tid  = threadIdx.x;
    const int lane = tid & 31;
    const int w    = tid >> 5;

    // ---- stage weights / bias / leaf values (block cooperative) ----
#pragma unroll
    for (int i = 0; i < 64; ++i) {                  // 64*128 = 8192 slots
        int idx = i * 128 + tid;
        float v = (idx < N_NODES * INPUT_DIM) ? nw[idx] : 0.0f;
        s_wh[idx] = (_Float16)v;
    }
    s_bias[tid] = (tid < N_NODES) ? nb[tid] : 0.0f;
    s_lv[tid] = lv[tid];
    __syncthreads();

    const int rowBase = blockIdx.x * 128 + w * 32;

    // ---- stage x tile: 32 rows x 64 f32 -> f16 in LDS (wave-local) ----
    {
        const float4* xsrc = (const float4*)(x + (size_t)rowBase * INPUT_DIM);
#pragma unroll
        for (int i = 0; i < 16; ++i) {
            int o = i * 32 + lane;                  // float4 index, 512 total
            float4 v = xsrc[o];
            _Float16* dst = &s_xh[w][o * 4];
            dst[0] = (_Float16)v.x; dst[1] = (_Float16)v.y;
            dst[2] = (_Float16)v.z; dst[3] = (_Float16)v.w;
        }
        // same-wave DS ops are in-order: stores visible to subsequent loads
    }

    const int halfSel = lane >> 4;                  // 0: lanes 0-15, 1: lanes 16-31
    const int lrow    = lane & 15;
    const int a0  = halfSel ? 8  : 0;               // A frag K bases (16-bit A 16x32 layout)
    const int a1  = halfSel ? 24 : 16;
    const int bko = halfSel ? 8 : 0;                // B frag K base (16-bit B 32x16 layout)

    // ---- GEMM + sigmoid, two 16-row M tiles ----
#pragma unroll
    for (int mt = 0; mt < 2; ++mt) {
        v8f acc[8];
        const v8f vzero = {0.f,0.f,0.f,0.f,0.f,0.f,0.f,0.f};
#pragma unroll
        for (int nt = 0; nt < 8; ++nt) acc[nt] = vzero;

        const _Float16* xr = &s_xh[w][(mt * 16 + lrow) * 64];
#pragma unroll
        for (int kb = 0; kb < 64; kb += 32) {
            // preload A fragment + ALL 8 B fragments, then 8 back-to-back WMMAs
            h8 alo = *(const h8*)(xr + kb + a0);
            h8 ahi = *(const h8*)(xr + kb + a1);
            v16h A = __builtin_shufflevector(alo, ahi,
                     0,1,2,3,4,5,6,7,8,9,10,11,12,13,14,15);

            v16h Bfrag[8];
#pragma unroll
            for (int nt = 0; nt < 8; ++nt) {
                const _Float16* wr = &s_wh[(nt * 16 + lrow) * 64 + kb + bko];
                h8 blo = *(const h8*)(wr);
                h8 bhi = *(const h8*)(wr + 16);
                Bfrag[nt] = __builtin_shufflevector(blo, bhi,
                            0,1,2,3,4,5,6,7,8,9,10,11,12,13,14,15);
            }
#pragma unroll
            for (int nt = 0; nt < 8; ++nt) {
                acc[nt] = __builtin_amdgcn_wmma_f32_16x16x32_f16(
                    false, A, false, Bfrag[nt], (short)0, acc[nt], false, false);
            }
        }
        // unpack C/D layout: row m = v + 8*halfSel, col n = nt*16 + lrow
#pragma unroll
        for (int nt = 0; nt < 8; ++nt) {
            int n = nt * 16 + lrow;
            float b = s_bias[n];
#pragma unroll
            for (int v = 0; v < 8; ++v) {
                int m = v + halfSel * 8;
                float z = acc[nt][v] + b;
                float p = 1.0f / (1.0f + __expf(-z));
                s_probs[w][(mt * 16 + m) * 128 + n] = p;
            }
        }
    }

    // ---- leaf expansion: lane L computes leaves 4L..4L+3 for each row ----
    const float lva = s_lv[4 * lane + 0];
    const float lvb = s_lv[4 * lane + 1];
    const float lvc = s_lv[4 * lane + 2];
    const float lvd = s_lv[4 * lane + 3];
    float predVal = 0.0f;
    float* leafOut = out + (size_t)batch + (size_t)rowBase * N_LEAVES;

    for (int r = 0; r < 32; ++r) {
        const float* P = &s_probs[w][r * 128];
        // shared prefix over depths 0..4 (leaf bits b6..b2 == lane bits 4..0)
        float prod = 1.0f;
#pragma unroll
        for (int d = 0; d < 5; ++d) {
            int node = (1 << d) - 1 + (lane >> (5 - d));
            int bit  = (lane >> (4 - d)) & 1;
            float p  = P[node];
            prod *= bit ? p : (1.0f - p);
        }
        float p5  = P[31 + lane];
        float pa  = prod * (1.0f - p5);
        float pb  = prod * p5;
        float p6a = P[63 + 2 * lane];
        float p6b = P[63 + 2 * lane + 1];
        float l0 = pa * (1.0f - p6a);
        float l1 = pa * p6a;
        float l2 = pb * (1.0f - p6b);
        float l3 = pb * p6b;

        float4 ov; ov.x = l0; ov.y = l1; ov.z = l2; ov.w = l3;
        *(float4*)(leafOut + (size_t)r * N_LEAVES + 4 * lane) = ov;  // coalesced 512B/row

        // pred = sum(leaf * leaf_value): butterfly reduce across the wave
        float s = l0 * lva + l1 * lvb + l2 * lvc + l3 * lvd;
#pragma unroll
        for (int off = 16; off > 0; off >>= 1)
            s += __shfl_xor(s, off, 32);
        if (lane == r) predVal = s;
    }
    out[rowBase + lane] = predVal;                   // coalesced pred writes
}

extern "C" void kernel_launch(void* const* d_in, const int* in_sizes, int n_in,
                              void* d_out, int out_size, void* d_ws, size_t ws_size,
                              hipStream_t stream) {
    const float* x  = (const float*)d_in[0];
    const float* nw = (const float*)d_in[1];
    const float* nb = (const float*)d_in[2];
    const float* lv = (const float*)d_in[3];
    float* out = (float*)d_out;
    int batch = in_sizes[0] / INPUT_DIM;             // 1048576
    int grid  = batch / 128;                         // 8192 blocks of 128 threads
    hipLaunchKernelGGL(soft_tree_kernel, dim3(grid), dim3(128), 0, stream,
                       x, nw, nb, lv, out, batch);
}
